// BiRNN_CRF_249108103771
// MI455X (gfx1250) — matrix operand order repaired and checked by reading, not compiled, not run on hardware
//
#include <hip/hip_runtime.h>
#include <hip/hip_bf16.h>
#include <cstdint>

typedef unsigned short ushort_t;
typedef __attribute__((ext_vector_type(16))) __bf16   v16bf;
typedef __attribute__((ext_vector_type(8)))  float    v8f;
typedef __attribute__((ext_vector_type(4)))  unsigned uv4;

#define BB 64
#define TT 512
#define EE 768
#define HH 512
#define KK 32
#define GH 2048        /* 4H */
#define CH 1024        /* 2H */
#define MROWS (BB*TT)  /* 32768 */

union FragBF { v16bf v; uv4 q[2]; };

__device__ __forceinline__ ushort_t f2bf(float x) {
  unsigned u = __float_as_uint(x);
  u += 0x7FFFu + ((u >> 16) & 1u);          // round-to-nearest-even
  return (ushort_t)(u >> 16);
}
__device__ __forceinline__ float bf2f(ushort_t h) {
  return __uint_as_float(((unsigned)h) << 16);
}

// A fragment (16x32 bf16, M x K): lane m=lane&15 holds two contiguous 8-elem
// chunks at K = ((lane>>4)<<3) and +16  (per ISA 7.12.2 16-bit A layout).
__device__ __forceinline__ v16bf load_frag_a(const ushort_t* base, size_t ld, int lane) {
  const ushort_t* r = base + (size_t)(lane & 15) * ld + (size_t)((lane >> 4) << 3);
  FragBF f;
  f.q[0] = *(const uv4*)(r);
  f.q[1] = *(const uv4*)(r + 16);
  return f.v;
}
// B fragment (32x16 bf16, K x N) sourced from row-major (N,K) weights:
// lane n=lane&15 holds two contiguous 8-elem chunks at K = ((lane>>4)<<3) and +16.
__device__ __forceinline__ v16bf load_frag_b(const ushort_t* base, size_t ld, int lane) {
  const ushort_t* r = base + (size_t)(lane & 15) * ld + (size_t)((lane >> 4) << 3);
  FragBF f;
  f.q[0] = *(const uv4*)(r);
  f.q[1] = *(const uv4*)(r + 16);
  return f.v;
}
__device__ __forceinline__ v8f wmma_bf16(v16bf a, v16bf b, v8f c) {
  return __builtin_amdgcn_wmma_f32_16x16x32_bf16(false, a, false, b, (short)0, c, false, false);
}
// C/D 16x16 f32: vgpr v -> m = v + ((lane>>4)<<3), n = lane&15
__device__ __forceinline__ v8f load_frag_c(const float* base, size_t ld, int lane) {
  const int n = lane & 15, mh = (lane >> 4) << 3;
  v8f c;
  #pragma unroll
  for (int v = 0; v < 8; ++v) c[v] = base[(size_t)(v + mh) * ld + n];
  return c;
}
__device__ __forceinline__ void store_frag_c(float* base, size_t ld, int lane, v8f c) {
  const int n = lane & 15, mh = (lane >> 4) << 3;
  #pragma unroll
  for (int v = 0; v < 8; ++v) base[(size_t)(v + mh) * ld + n] = c[v];
}

// CDNA5 async global->LDS copy (16B per lane), tracked by ASYNCcnt.
// VFLAT async form: vdst = LDS byte address, vaddr = 64-bit global address.
__device__ __forceinline__ void async_g2l_b128(void* lds, const void* gptr) {
#if defined(__AMDGCN__)
  asm volatile("global_load_async_to_lds_b128 %0, %1, off"
               :: "v"((unsigned)(uintptr_t)lds),
                  "v"((unsigned long long)(uintptr_t)gptr)
               : "memory");
#endif
}
__device__ __forceinline__ void wait_async0() {
#if defined(__AMDGCN__)
  asm volatile("s_wait_asynccnt 0x0" ::: "memory");
#endif
}

// ---------------- generic fp32 -> bf16 convert ----------------
__global__ void k_cvt(const float* __restrict__ s, ushort_t* __restrict__ d, int n) {
  int i = blockIdx.x * blockDim.x + threadIdx.x;
  const int stride = gridDim.x * blockDim.x;
  for (; i < n; i += stride) d[i] = f2bf(s[i]);
}

// ---------------- init (graph-replay safe) ----------------
__global__ void k_zero(float* out, unsigned* ctrs) {
  if (threadIdx.x == 0 && blockIdx.x == 0) { out[0] = 0.f; ctrs[0] = 0u; ctrs[1] = 0u; }
}

// ---------------- pre = x @ w_ih^T + b_ih + b_hh  (M=32768,N=2048,K=768) ----
__global__ void __launch_bounds__(256) k_gemm_pre(
    const ushort_t* __restrict__ A,   // M x 768 bf16
    const ushort_t* __restrict__ W,   // 2048 x 768 bf16 (N,K row-major)
    const float* __restrict__ b1, const float* __restrict__ b2,
    float* __restrict__ C)            // M x 2048 f32
{
  const int lane = threadIdx.x & 31;
  const int wave = threadIdx.x >> 5;
  const int m0 = blockIdx.x * 16;
  const int n0 = blockIdx.y * 128 + wave * 16;
  const float bv = b1[n0 + (lane & 15)] + b2[n0 + (lane & 15)];
  v8f acc;
  #pragma unroll
  for (int v = 0; v < 8; ++v) acc[v] = bv;
  const ushort_t* pa = A + (size_t)m0 * EE;
  const ushort_t* pw = W + (size_t)n0 * EE;
  for (int k0 = 0; k0 < EE; k0 += 32) {
    __builtin_prefetch(pa + (size_t)(lane & 15) * EE + k0 + 128, 0, 1);
    v16bf af = load_frag_a(pa + k0, EE, lane);
    v16bf bf = load_frag_b(pw + k0, EE, lane);
    acc = wmma_bf16(af, bf, acc);
  }
  store_frag_c(C + (size_t)m0 * GH + n0, GH, lane, acc);
}

// ---------------- BiLSTM recurrence ----------------
// grid = 16 blocks: dir = blk>>3 (0=fwd,1=bwd); each block owns 64 hidden units.
// Per step: gates(64x256) = pre_tile + h_{t-1}(64x512) @ w_hh_slice^T via WMMA.
// h_{t-1} staged into LDS with GLOBAL_LOAD_ASYNC_TO_LDS_B128 (ASYNCcnt path);
// the gate buffer aliases the stage buffer after a barrier.
// Cross-block sync: global atomic arrive counter + spin (8 blocks per dir).
__global__ void __launch_bounds__(512) k_lstm(
    const float* __restrict__ pre_f, const float* __restrict__ pre_b,
    const ushort_t* __restrict__ whh_f, const ushort_t* __restrict__ whh_b,
    ushort_t* __restrict__ hprev,   // [dir*2+buf] x (64*512) bf16 ping-pong
    ushort_t* __restrict__ hbf,     // (B*T) x 1024 bf16, cols [f(512) | b(512)]
    unsigned* __restrict__ ctr)
{
  extern __shared__ char smem[];
  ushort_t* hlds = (ushort_t*)smem;   // 64 x 512 bf16 (64KB)
  float*    gbuf = (float*)smem;      // aliased after barrier: [256 col][64 b] f32

  const int dir  = (int)(blockIdx.x >> 3);
  const int blk  = (int)(blockIdx.x & 7);
  const int tid  = threadIdx.x;
  const int lane = tid & 31;
  const int wave = tid >> 5;
  const float*    pre = dir ? pre_b : pre_f;
  const ushort_t* whh = dir ? whh_b : whh_f;
  unsigned* cnt = ctr + dir;
  const int jblk = blk * 64;
  const int gate = wave >> 2;                       // 0:i 1:f 2:g 3:o
  const int nrow = gate * HH + jblk + (wave & 3) * 16; // gate-col base (of 2048)
  const int ab  = tid & 63;                         // batch for activation
  const int aj0 = (tid >> 6) * 8;                   // 8 hidden units / thread
  float cst[8];
  #pragma unroll
  for (int i = 0; i < 8; ++i) cst[i] = 0.f;

  for (int t = 0; t < TT; ++t) {
    const int tt = dir ? (TT - 1 - t) : t;
    // stage h_{t-1} (or zeros) into LDS
    if (t == 0) {
      uv4 z = {0u, 0u, 0u, 0u};
      for (int i = tid; i < 4096; i += 512) ((uv4*)hlds)[i] = z;
    } else {
      const ushort_t* src = hprev + (size_t)(dir * 2 + (t & 1)) * (BB * HH);
      #pragma unroll
      for (int r = 0; r < 8; ++r) {
        const int i = tid + r * 512;                // uv4 (16B) index
        async_g2l_b128(hlds + (size_t)i * 8, src + (size_t)i * 8);
      }
      wait_async0();
    }
    __syncthreads();

    // accumulators initialized from precomputed input projection
    v8f acc[4];
    #pragma unroll
    for (int mt = 0; mt < 4; ++mt)
      acc[mt] = load_frag_c(pre + ((size_t)(mt * 16) * TT + tt) * GH + nrow,
                            (size_t)TT * GH, lane);
    for (int k0 = 0; k0 < HH; k0 += 32) {
      v16bf bf = load_frag_b(whh + (size_t)nrow * HH + k0, HH, lane);
      #pragma unroll
      for (int mt = 0; mt < 4; ++mt) {
        v16bf af = load_frag_a(hlds + (size_t)(mt * 16) * HH + k0, HH, lane);
        acc[mt] = wmma_bf16(af, bf, acc[mt]);
      }
    }
    __syncthreads();   // all waves done reading hlds; reuse LDS as gate buffer

    {
      const int lc = wave * 16 + (lane & 15);       // == gate*64 + j_local
      const int mh = (lane >> 4) << 3;
      #pragma unroll
      for (int mt = 0; mt < 4; ++mt) {
        #pragma unroll
        for (int v = 0; v < 8; ++v)
          gbuf[(size_t)lc * 64 + mt * 16 + mh + v] = acc[mt][v];
      }
    }
    __syncthreads();

    // pointwise LSTM cell: c = f*c + i*g ; h = o*tanh(c)
    unsigned pk[4];
    #pragma unroll
    for (int u2 = 0; u2 < 4; ++u2) {
      float h2[2];
      #pragma unroll
      for (int s = 0; s < 2; ++s) {
        const int u = u2 * 2 + s;
        const int j = aj0 + u;
        float ig = gbuf[(size_t)(0 * 64 + j) * 64 + ab];
        float fg = gbuf[(size_t)(1 * 64 + j) * 64 + ab];
        float gg = gbuf[(size_t)(2 * 64 + j) * 64 + ab];
        float og = gbuf[(size_t)(3 * 64 + j) * 64 + ab];
        ig = 1.f / (1.f + __expf(-ig));
        fg = 1.f / (1.f + __expf(-fg));
        og = 1.f / (1.f + __expf(-og));
        gg = tanhf(gg);
        float c = fg * cst[u] + ig * gg;
        cst[u] = c;
        h2[s] = og * tanhf(c);
      }
      pk[u2] = (unsigned)f2bf(h2[0]) | ((unsigned)f2bf(h2[1]) << 16);
    }
    uv4 hv = { pk[0], pk[1], pk[2], pk[3] };
    *(uv4*)(hprev + (size_t)(dir * 2 + ((t + 1) & 1)) * (BB * HH)
            + (size_t)ab * HH + jblk + aj0) = hv;
    *(uv4*)(hbf + ((size_t)ab * TT + tt) * CH + dir * HH + jblk + aj0) = hv;

    // cross-block barrier for this direction (8 co-resident blocks)
    __threadfence();
    __syncthreads();
    if (tid == 0) {
      atomicAdd(cnt, 1u);
      const unsigned tgt = (unsigned)(8 * (t + 1));
      while (__hip_atomic_load(cnt, __ATOMIC_ACQUIRE, __HIP_MEMORY_SCOPE_AGENT) < tgt)
        __builtin_amdgcn_s_sleep(2);
    }
    __syncthreads();
  }
}

// ---------------- attention scores: s[b,t] = h . w_attn + b_attn ----------------
__global__ void __launch_bounds__(256) k_scores(
    const ushort_t* __restrict__ hbf, const float* __restrict__ wattn,
    const float* __restrict__ battn, const int* __restrict__ amask,
    float* __restrict__ scores)
{
  const int row  = blockIdx.x * 8 + (threadIdx.x >> 5);
  const int lane = threadIdx.x & 31;
  float s = 0.f;
  for (int k = lane; k < CH; k += 32)
    s += bf2f(hbf[(size_t)row * CH + k]) * wattn[k];
  #pragma unroll
  for (int o = 16; o > 0; o >>= 1) s += __shfl_xor(s, o, 32);
  if (lane == 0)
    scores[row] = (amask[row] == 0) ? -1.0e9f : (s + battn[0]);
}

// ---------------- softmax over t (per batch) in-place ----------------
__global__ void __launch_bounds__(256) k_softmax(float* __restrict__ scores) {
  const int b = blockIdx.x, tid = threadIdx.x;
  __shared__ float red[256];
  float m = -1.0e30f;
  for (int t = tid; t < TT; t += 256) m = fmaxf(m, scores[b * TT + t]);
  red[tid] = m; __syncthreads();
  for (int s = 128; s > 0; s >>= 1) { if (tid < s) red[tid] = fmaxf(red[tid], red[tid + s]); __syncthreads(); }
  m = red[0]; __syncthreads();
  float sum = 0.f;
  for (int t = tid; t < TT; t += 256) sum += __expf(scores[b * TT + t] - m);
  red[tid] = sum; __syncthreads();
  for (int s = 128; s > 0; s >>= 1) { if (tid < s) red[tid] += red[tid + s]; __syncthreads(); }
  const float inv = 1.f / red[0]; __syncthreads();
  for (int t = tid; t < TT; t += 256) scores[b * TT + t] = __expf(scores[b * TT + t] - m) * inv;
}

// ---------------- context[b,h] = sum_t attn[b,t] * h[b,t,h] ----------------
__global__ void __launch_bounds__(256) k_context(
    const float* __restrict__ attn, const ushort_t* __restrict__ hbf,
    float* __restrict__ ctx)
{
  const int b = blockIdx.x;
  const int n = blockIdx.y * 256 + threadIdx.x;
  float s = 0.f;
  for (int t = 0; t < TT; ++t)
    s += attn[b * TT + t] * bf2f(hbf[((size_t)b * TT + t) * CH + n]);
  ctx[(size_t)b * CH + n] = s;
}

// ---------------- ctxk[b,k] = context[b] . w_tag[k,2H:4H] + b_tag[k] ----------------
__global__ void __launch_bounds__(256) k_ctxk(
    const float* __restrict__ ctx, const float* __restrict__ wtag,
    const float* __restrict__ btag, float* __restrict__ ctxk)
{
  const int idx  = blockIdx.x * 8 + (threadIdx.x >> 5);  // b*32+k, 2048 total
  const int lane = threadIdx.x & 31;
  const int b = idx >> 5, k = idx & 31;
  float s = 0.f;
  for (int h = lane; h < CH; h += 32)
    s += ctx[(size_t)b * CH + h] * wtag[(size_t)k * GH + CH + h];
  #pragma unroll
  for (int o = 16; o > 0; o >>= 1) s += __shfl_xor(s, o, 32);
  if (lane == 0) ctxk[idx] = s + btag[k];
}

// ---------------- emissions = h . w_tag[:, :2H]^T + ctxk  (M=32768,N=32,K=1024) ----
__global__ void __launch_bounds__(256) k_emis(
    const ushort_t* __restrict__ hbf, const ushort_t* __restrict__ wtag,
    const float* __restrict__ ctxk, float* __restrict__ emis)
{
  const int lane = threadIdx.x & 31;
  const int wave = threadIdx.x >> 5;
  const int m0 = (blockIdx.x * 8 + wave) * 16;
  v8f a0, a1;
  #pragma unroll
  for (int v = 0; v < 8; ++v) { a0[v] = 0.f; a1[v] = 0.f; }
  const ushort_t* pa = hbf + (size_t)m0 * CH;
  for (int k0 = 0; k0 < CH; k0 += 32) {
    v16bf af = load_frag_a(pa + k0, CH, lane);
    v16bf b0 = load_frag_b(wtag + k0, GH, lane);                     // tags 0..15
    v16bf b1 = load_frag_b(wtag + (size_t)16 * GH + k0, GH, lane);   // tags 16..31
    a0 = wmma_bf16(af, b0, a0);
    a1 = wmma_bf16(af, b1, a1);
  }
  const int n = lane & 15, mh = (lane >> 4) << 3;
  #pragma unroll
  for (int v = 0; v < 8; ++v) {
    const int m = m0 + mh + v;
    const int b = m >> 9;                 // row = b*T + t, T = 512
    emis[(size_t)m * KK + n]      = a0[v] + ctxk[b * KK + n];
    emis[(size_t)m * KK + 16 + n] = a1[v] + ctxk[b * KK + 16 + n];
  }
}

// ---------------- CRF NLL: one wave per batch, lane = tag state ----------------
__global__ void __launch_bounds__(32) k_crf(
    const float* __restrict__ emis, const int* __restrict__ tags,
    const int* __restrict__ amask, const float* __restrict__ trans,
    const float* __restrict__ strans, const float* __restrict__ etrans,
    float* __restrict__ out)
{
  const int b = blockIdx.x, j = threadIdx.x;   // j in [0,32)
  __shared__ float lt[KK * KK];
  for (int i = j; i < KK * KK; i += 32) lt[i] = trans[i];
  __syncthreads();
  const float* eb = emis + (size_t)b * TT * KK;
  const int*   tb = tags + b * TT;
  const int*   mb = amask + b * TT;

  // numerator (lane-parallel over t)
  float nump = 0.f; int cnt = 0;
  for (int t = j; t < TT; t += 32) {
    const float mf = (float)mb[t];
    cnt += (mb[t] != 0) ? 1 : 0;
    nump += eb[t * KK + tb[t]] * mf;
    if (t >= 1) nump += lt[tb[t - 1] * KK + tb[t]] * mf;
  }
  #pragma unroll
  for (int o = 16; o > 0; o >>= 1) { nump += __shfl_xor(nump, o, 32); cnt += __shfl_xor(cnt, o, 32); }

  // forward algorithm: alpha[j] per lane
  float alpha = strans[j] + eb[j];
  for (int t = 1; t < TT; ++t) {
    float mx = -1.0e30f, s = 0.f;
    #pragma unroll
    for (int i = 0; i < KK; ++i) {
      const float v = __shfl(alpha, i, 32) + lt[i * KK + j];
      const float nm = fmaxf(mx, v);
      s = s * __expf(mx - nm) + __expf(v - nm);
      mx = nm;
    }
    const float nw = mx + __logf(s) + eb[t * KK + j];
    alpha = (mb[t] != 0) ? nw : alpha;
  }
  float v = alpha + etrans[j];
  float mx = v;
  #pragma unroll
  for (int o = 16; o > 0; o >>= 1) mx = fmaxf(mx, __shfl_xor(mx, o, 32));
  float s = __expf(v - mx);
  #pragma unroll
  for (int o = 16; o > 0; o >>= 1) s += __shfl_xor(s, o, 32);
  const float logZ = mx + __logf(s);
  if (j == 0) {
    float num = nump + strans[tb[0]];
    const int last = cnt - 1;
    num += etrans[tb[last]];
    atomicAdd(out, (logZ - num) * (1.f / (float)BB));
  }
}

// =====================================================================
extern "C" void kernel_launch(void* const* d_in, const int* in_sizes, int n_in,
                              void* d_out, int out_size, void* d_ws, size_t ws_size,
                              hipStream_t stream) {
  const float* x      = (const float*)d_in[0];
  const int*   tags   = (const int*)d_in[1];
  const int*   amask  = (const int*)d_in[2];
  const float* wihf   = (const float*)d_in[3];
  const float* whhf   = (const float*)d_in[4];
  const float* bihf   = (const float*)d_in[5];
  const float* bhhf   = (const float*)d_in[6];
  const float* wihb   = (const float*)d_in[7];
  const float* whhb   = (const float*)d_in[8];
  const float* bihb   = (const float*)d_in[9];
  const float* bhhb   = (const float*)d_in[10];
  const float* wattn  = (const float*)d_in[11];
  const float* battn  = (const float*)d_in[12];
  const float* wtag   = (const float*)d_in[13];
  const float* btag   = (const float*)d_in[14];
  const float* trans  = (const float*)d_in[15];
  const float* strans = (const float*)d_in[16];
  const float* etrans = (const float*)d_in[17];
  float* out = (float*)d_out;

  char* ws = (char*)d_ws;
  size_t off = 0;
  auto alloc = [&](size_t bytes) -> void* {
    void* p = ws + off;
    off += (bytes + 255) & ~(size_t)255;
    return p;
  };
  ushort_t* XB    = (ushort_t*)alloc((size_t)MROWS * EE * 2);   // x bf16
  ushort_t* WIHFB = (ushort_t*)alloc((size_t)GH * EE * 2);
  ushort_t* WIHBB = (ushort_t*)alloc((size_t)GH * EE * 2);
  ushort_t* WHHFB = (ushort_t*)alloc((size_t)GH * HH * 2);
  ushort_t* WHHBB = (ushort_t*)alloc((size_t)GH * HH * 2);
  ushort_t* WTAGB = (ushort_t*)alloc((size_t)KK * GH * 2);
  float*    PREF  = (float*)   alloc((size_t)MROWS * GH * 4);
  float*    PREB  = (float*)   alloc((size_t)MROWS * GH * 4);
  ushort_t* HBF   = (ushort_t*)alloc((size_t)MROWS * CH * 2);   // concat h bf16
  ushort_t* HPREV = (ushort_t*)alloc((size_t)4 * BB * HH * 2);  // ping-pong x 2 dirs
  unsigned* CTRS  = (unsigned*)alloc(256);
  float*    SCORE = (float*)   alloc((size_t)MROWS * 4);
  float*    CTX   = (float*)   alloc((size_t)BB * CH * 4);
  float*    CTXK  = (float*)   alloc((size_t)BB * KK * 4);
  float*    EMIS  = (float*)   alloc((size_t)MROWS * KK * 4);

  // 1) converts to bf16
  k_cvt<<<2048, 256, 0, stream>>>(x,    XB,    MROWS * EE);
  k_cvt<<<2048, 256, 0, stream>>>(wihf, WIHFB, GH * EE);
  k_cvt<<<2048, 256, 0, stream>>>(wihb, WIHBB, GH * EE);
  k_cvt<<<2048, 256, 0, stream>>>(whhf, WHHFB, GH * HH);
  k_cvt<<<2048, 256, 0, stream>>>(whhb, WHHBB, GH * HH);
  k_cvt<<<256,  256, 0, stream>>>(wtag, WTAGB, KK * GH);

  // 2) input projections (WMMA GEMM)
  dim3 gp(MROWS / 16, GH / 128);
  k_gemm_pre<<<gp, 256, 0, stream>>>(XB, WIHFB, bihf, bhhf, PREF);
  k_gemm_pre<<<gp, 256, 0, stream>>>(XB, WIHBB, bihb, bhhb, PREB);

  // 3) init output + barrier counters, then BiLSTM recurrence (WMMA + async LDS)
  k_zero<<<1, 32, 0, stream>>>(out, CTRS);
  k_lstm<<<16, 512, 65536, stream>>>(PREF, PREB, WHHFB, WHHBB, HPREV, HBF, CTRS);

  // 4) attention + context
  k_scores<<<MROWS / 8, 256, 0, stream>>>(HBF, wattn, battn, amask, SCORE);
  k_softmax<<<BB, 256, 0, stream>>>(SCORE);
  k_context<<<dim3(BB, CH / 256), 256, 0, stream>>>(SCORE, HBF, CTX);
  k_ctxk<<<(BB * KK) / 8, 256, 0, stream>>>(CTX, wtag, btag, CTXK);

  // 5) emissions (WMMA) + CRF NLL
  k_emis<<<MROWS / 128, 256, 0, stream>>>(HBF, WTAGB, CTXK, EMIS);
  k_crf<<<BB, 32, 0, stream>>>(EMIS, tags, amask, trans, strans, etrans, out);
}
